// pLayer_64201171140784
// MI455X (gfx1250) — matrix operand order repaired and checked by reading, not compiled, not run on hardware
//
#include <hip/hip_runtime.h>
#include <math.h>

typedef __attribute__((ext_vector_type(8)))  _Float16 v8h;
typedef __attribute__((ext_vector_type(16))) _Float16 v16h;
typedef __attribute__((ext_vector_type(8)))  float    v8f;

#define GMIN_C 0.01f
#define GMAX_C 10.0f
#define PGMIN_C 0.1f
#define EPSTR_C 0.1f

static constexpr int NIN = 1024, NOUT = 1024, MDIM = 1026, MPAD = 1056;
static constexpr int FN = 4, VN = 8, EN = 256, FV = 32;

// workspace layout (float offsets)
static constexpr int WS_COLMIN = 0;       // 1024
static constexpr int WS_COLSUM = 1024;    // 1024
static constexpr int WS_ROWSUM = 2048;    // 1056 (zero-padded)
static constexpr int WS_INVG   = 3104;    // 32*1024
static constexpr int WS_PART   = 35872;   // 3*512
static constexpr int NWG_GEMM  = 512;

// ---------- per-column stats: colmin = min_m |theta|, colsum_gt = 0.1*sum_m|theta|/colmin
__global__ void kColStats(const float* __restrict__ theta, float* __restrict__ ws) {
  int n = blockIdx.x * blockDim.x + threadIdx.x;
  if (n >= NOUT) return;
  float mn = 1e30f, sm = 0.f;
  for (int m = 0; m < MDIM; ++m) {
    float t = fabsf(theta[(size_t)m * NOUT + n]);
    mn = fminf(mn, t);
    sm += t;
  }
  ws[WS_COLMIN + n] = mn;
  ws[WS_COLSUM + n] = PGMIN_C * sm / mn;
}

// ---------- rowsum_gt[m] = sum_n |theta[m,n]| * 0.1/colmin[n]  (zero for padded rows)
__global__ void kRowSum(const float* __restrict__ theta, float* __restrict__ ws) {
  int wave = threadIdx.x >> 5, lane = threadIdx.x & 31;
  int row = blockIdx.x * 8 + wave;
  if (row >= MPAD) return;
  float s = 0.f;
  if (row < MDIM) {
    for (int c = lane; c < NOUT; c += 32)
      s += fabsf(theta[(size_t)row * NOUT + c]) * (PGMIN_C / ws[WS_COLMIN + c]);
  }
  for (int o = 16; o > 0; o >>= 1) s += __shfl_xor(s, o, 32);
  if (lane == 0) ws[WS_ROWSUM + row] = s;
}

// ---------- invG[fv,n] = 1/(sum_m |theta_eff*noise| + 1e-10)
__global__ void kInvG(const float* __restrict__ theta, const float* __restrict__ noise,
                      float* __restrict__ ws) {
  int gid = blockIdx.x * blockDim.x + threadIdx.x;  // 32768
  int fv = gid >> 10, n = gid & 1023;
  const float* np = noise + (size_t)fv * MDIM * NOUT + n;
  float G = 0.f;
  for (int m = 0; m < MDIM; ++m) {
    float t = theta[(size_t)m * NOUT + n];
    t = fminf(fmaxf(t, -GMAX_C), GMAX_C);
    if (fabsf(t) < GMIN_C) t = 0.f;
    G += fabsf(t * fmaf(np[(size_t)m * NOUT], 0.2f, 0.9f));
  }
  ws[WS_INVG + gid] = 1.f / (G + 1e-10f);
}

// ---------- fused dual-GEMM + epilogue ----------
__global__ void __launch_bounds__(256)
kGemm(const float* __restrict__ a, const float* __restrict__ theta,
      const float* __restrict__ noise, const float* __restrict__ eta,
      const float* __restrict__ ws, float* __restrict__ part,
      float* __restrict__ out) {
  __shared__ __align__(16) _Float16 Xs[128][40];  // [e][k], padded stride (80B rows)
  __shared__ __align__(16) _Float16 Wt[128][40];  // [n][k]  (B operand of z-GEMM)
  __shared__ __align__(16) _Float16 Bt[128][40];  // [n][k]  (B operand of q-GEMM)
  __shared__ float red[3][8];

  const int tid = threadIdx.x;
  const int wave = tid >> 5, lane = tid & 31;
  const int hs = lane >> 4, l16 = lane & 15;
  const int fv = blockIdx.z;
  const int eBase = blockIdx.y * 128;
  const int nBase = blockIdx.x * 128;
  const int waveE = wave >> 2, waveN = wave & 3;
  const bool doT1 = (blockIdx.x == 0);

  const float et0 = eta[0], et1 = eta[1], et2 = eta[2], et3 = eta[3];

  const v8f zf = {};
  v8f accZ[4][2], accQ[4][2];
#pragma unroll
  for (int i = 0; i < 4; ++i)
#pragma unroll
    for (int j = 0; j < 2; ++j) { accZ[i][j] = zf; accQ[i][j] = zf; }

  float t1p = 0.f;
  const float* rs = ws + WS_ROWSUM;

  // ---- X-fill mapping: thread -> (row xe, k-half xh); 16 contiguous k per thread
  const int xe = tid >> 1;
  const int xh = (tid & 1) * 16;
  const float* aRow = a + ((size_t)fv * EN + eBase + xe) * NIN;

  // ---- W/B-fill mapping: thread -> fixed column n, k-half wk; all n-invariants hoisted
  const int wn = tid & 127;
  const int wk = (tid >> 7) * 16;
  const float ign = ws[WS_INVG + fv * NOUT + nBase + wn];
  const float gsc = PGMIN_C / ws[WS_COLMIN + nBase + wn];
  const float* thCol = theta + nBase + wn;
  const float* nzCol = noise + (size_t)fv * MDIM * NOUT + nBase + wn;

  auto fillWB = [&](int mb, bool tail) {
    union { _Float16 h[16]; v8h v[2]; } wu, bu;
#pragma unroll
    for (int i = 0; i < 16; ++i) {
      float w = 0.f, b = 0.f;
      if (!tail || (mb + i) < MDIM) {
        size_t off = (size_t)(mb + i) * NOUT;
        float th = thCol[off];
        float u  = nzCol[off];
        float te = fminf(fmaxf(th, -GMAX_C), GMAX_C);
        te = (fabsf(te) < GMIN_C) ? 0.f : te;
        float tn = te * fmaf(u, 0.2f, 0.9f);
        w = tn * ign;
        b = copysignf(fabsf(th) * gsc, tn);
      }
      wu.h[i] = (_Float16)w;
      bu.h[i] = (_Float16)b;
    }
    *(v8h*)&Wt[wn][wk]     = wu.v[0];
    *(v8h*)&Wt[wn][wk + 8] = wu.v[1];
    *(v8h*)&Bt[wn][wk]     = bu.v[0];
    *(v8h*)&Bt[wn][wk + 8] = bu.v[1];
  };

  for (int kk = 0; kk < 33; ++kk) {
    if (kk < 32) {
      // ---- main fills: no bounds checks (all m < 1024 here)
      const int m0 = kk * 32 + xh;
      float xv[16];
      const float4* ap = (const float4*)(aRow + m0);
      *(float4*)&xv[0]  = ap[0];
      *(float4*)&xv[4]  = ap[1];
      *(float4*)&xv[8]  = ap[2];
      *(float4*)&xv[12] = ap[3];
      union { _Float16 h[16]; v8h v[2]; } xu;
#pragma unroll
      for (int j = 0; j < 16; ++j) xu.h[j] = (_Float16)xv[j];
      *(v8h*)&Xs[xe][xh]     = xu.v[0];
      *(v8h*)&Xs[xe][xh + 8] = xu.v[1];
      if (doT1) {  // term1 fused: each (fv,e,m) touched exactly once by n-tile 0
        float rv[16];
        const float4* rp = (const float4*)(rs + m0);
        *(float4*)&rv[0]  = rp[0];
        *(float4*)&rv[4]  = rp[1];
        *(float4*)&rv[8]  = rp[2];
        *(float4*)&rv[12] = rp[3];
#pragma unroll
        for (int j = 0; j < 16; ++j) t1p += xv[j] * xv[j] * rv[j];
      }
      fillWB(kk * 32 + wk, false);
      if (kk < 31)  // hint next noise K-chunk toward caches
        __builtin_prefetch(nzCol + (size_t)((kk + 1) * 32 + wk) * NOUT, 0, 1);
    } else {
      // ---- tail K-chunk: rows 1024 (ones), 1025 (zeros), pad to 1056
      union { _Float16 h[16]; v8h v[2]; } xu;
#pragma unroll
      for (int j = 0; j < 16; ++j) xu.h[j] = (_Float16)0.f;
      if (xh == 0) xu.h[0] = (_Float16)1.f;  // m == 1024 -> x = 1
      *(v8h*)&Xs[xe][xh]     = xu.v[0];
      *(v8h*)&Xs[xe][xh + 8] = xu.v[1];
      if (doT1 && xh == 0) t1p += rs[NIN];   // 1^2 * rowsum_gt[1024]
      fillWB(1024 + wk, true);
    }
    __syncthreads();

    // ---- fragments in contiguous k order, K = hs*16 + s (wave32, 16x16x32 f16)
    v16h aF[4], wF[2], bF[2];
#pragma unroll
    for (int t = 0; t < 4; ++t) {
      int row = waveE * 64 + t * 16 + l16;
      v8h lo = *(const v8h*)&Xs[row][hs * 16];      // K = hs*16 + 0..7
      v8h hi = *(const v8h*)&Xs[row][hs * 16 + 8];  // K = hs*16 + 8..15
      aF[t] = __builtin_shufflevector(lo, hi, 0, 1, 2, 3, 4, 5, 6, 7, 8, 9, 10, 11, 12, 13, 14, 15);
    }
#pragma unroll
    for (int t = 0; t < 2; ++t) {
      int col = waveN * 32 + t * 16 + l16;
      v8h lo = *(const v8h*)&Wt[col][hs * 16];      // K = hs*16 + 0..7
      v8h hi = *(const v8h*)&Wt[col][hs * 16 + 8];  // K = hs*16 + 8..15
      wF[t] = __builtin_shufflevector(lo, hi, 0, 1, 2, 3, 4, 5, 6, 7, 8, 9, 10, 11, 12, 13, 14, 15);
      lo = *(const v8h*)&Bt[col][hs * 16];
      hi = *(const v8h*)&Bt[col][hs * 16 + 8];
      bF[t] = __builtin_shufflevector(lo, hi, 0, 1, 2, 3, 4, 5, 6, 7, 8, 9, 10, 11, 12, 13, 14, 15);
    }
#pragma unroll
    for (int mi = 0; mi < 4; ++mi)
#pragma unroll
      for (int ni = 0; ni < 2; ++ni) {
        accZ[mi][ni] = __builtin_amdgcn_wmma_f32_16x16x32_f16(
            false, aF[mi], false, wF[ni], (short)0, accZ[mi][ni], false, false);
        accQ[mi][ni] = __builtin_amdgcn_wmma_f32_16x16x32_f16(
            false, aF[mi], false, bF[ni], (short)0, accQ[mi][ni], false, false);
      }
    __syncthreads();
  }

  // ---- epilogue: activation store, term2 (q*z), term3 (colsum_gt * z^2)
  float t2p = 0.f, t3p = 0.f;
  const float* cs = ws + WS_COLSUM;
#pragma unroll
  for (int ni = 0; ni < 2; ++ni) {
    int n = nBase + waveN * 32 + ni * 16 + l16;
    float cg = cs[n];
#pragma unroll
    for (int mi = 0; mi < 4; ++mi) {
      int erow = eBase + waveE * 64 + mi * 16 + hs * 8;  // C layout: M = r + 8*halfsel
#pragma unroll
      for (int r = 0; r < 8; ++r) {
        float z = accZ[mi][ni][r];
        float q = accQ[mi][ni][r];
        out[((size_t)fv * EN + (erow + r)) * NOUT + n] = et0 + et1 * tanhf((z - et2) * et3);
        t3p += cg * z * z;
        t2p += q * z;
      }
    }
  }

  // ---- deterministic partial reduction (no float atomics)
  for (int o = 16; o > 0; o >>= 1) {
    t1p += __shfl_xor(t1p, o, 32);
    t2p += __shfl_xor(t2p, o, 32);
    t3p += __shfl_xor(t3p, o, 32);
  }
  if (lane == 0) { red[0][wave] = t1p; red[1][wave] = t2p; red[2][wave] = t3p; }
  __syncthreads();
  if (tid == 0) {
    float s1 = 0.f, s2 = 0.f, s3 = 0.f;
    for (int w = 0; w < 8; ++w) { s1 += red[0][w]; s2 += red[1][w]; s3 += red[2][w]; }
    int wg = (blockIdx.z * gridDim.y + blockIdx.y) * gridDim.x + blockIdx.x;
    part[0 * NWG_GEMM + wg] = s1;
    part[1 * NWG_GEMM + wg] = s2;
    part[2 * NWG_GEMM + wg] = s3;
  }
}

// ---------- final scalar: power = (t1 - 2*t2 + t3) / (F*V*E)
__global__ void kFinal(const float* __restrict__ part, float* __restrict__ out) {
  if (threadIdx.x == 0 && blockIdx.x == 0) {
    float s1 = 0.f, s2 = 0.f, s3 = 0.f;
    for (int i = 0; i < NWG_GEMM; ++i) {
      s1 += part[i];
      s2 += part[NWG_GEMM + i];
      s3 += part[2 * NWG_GEMM + i];
    }
    out[(size_t)FV * EN * NOUT] = (s1 - 2.f * s2 + s3) * (1.f / (float)(FN * VN * EN));
  }
}

extern "C" void kernel_launch(void* const* d_in, const int* in_sizes, int n_in,
                              void* d_out, int out_size, void* d_ws, size_t ws_size,
                              hipStream_t stream) {
  const float* a     = (const float*)d_in[0];
  const float* theta = (const float*)d_in[1];
  const float* noise = (const float*)d_in[2];
  const float* eta   = (const float*)d_in[3];
  float* ws  = (float*)d_ws;
  float* out = (float*)d_out;

  kColStats<<<4, 256, 0, stream>>>(theta, ws);
  kRowSum<<<132, 256, 0, stream>>>(theta, ws);
  kInvG<<<128, 256, 0, stream>>>(theta, noise, ws);
  dim3 g(8, 2, 32);  // n-tiles x e-tiles x (f,v) batches
  kGemm<<<g, 256, 0, stream>>>(a, theta, noise, eta, ws, ws + WS_PART, out);
  kFinal<<<1, 32, 0, stream>>>(ws + WS_PART, out);
}
